// SGC_LSTM_50259707298503
// MI455X (gfx1250) — matrix operand order repaired and checked by reading, not compiled, not run on hardware
//
#include <hip/hip_runtime.h>

// ---------------------------------------------------------------------------
// SGCN + per-player LSTM for MI455X (gfx1250, wave32, WMMA 16x16x32 f16)
//
// Shapes: B=256 T=64 P=10 D=128 HID=64.
// Prep kernels: swizzle all weights to f16 in WMMA B-fragment order
//   ([frag][lane][16] contiguous) so each B operand is one 32B vector load.
// Kernel 1 (SGCN): one wave per (b,t) instance, P=10 padded to M=16.
//   base:  [16,256] @ [256,32]  (8 k-tiles x 2 n-tiles) per sign
//   deep:  [16,224] @ [224,32]  (7 k-tiles x 2 n-tiles) per sign, 2 layers
// Kernel 2 (LSTM): one block (4 waves) per (player, 16-batch tile);
//   16 WMMAs per step, h recurrence in LDS f16, c state in D-layout VGPRs.
// Activations use the CDNA5 hardware V_TANH_F32 (TRANS32, co-executes with
// WMMA) instead of branchy libm tanh/exp.
// ---------------------------------------------------------------------------

typedef __attribute__((ext_vector_type(16))) _Float16 v16h;
typedef __attribute__((ext_vector_type(4)))  _Float16 v4h;
typedef __attribute__((ext_vector_type(8)))  float    v8f;

#define BGAMES 256
#define TSTEPS 64
#define PNUM   10
#define DFEAT  128
#define HIDN   64
#define EPSV   1e-8f
#define K1_WAVES 4

// workspace layout (f16 elements)
#define Z_SZ   ((size_t)BGAMES * TSTEPS * PNUM * 64)   // 10,485,760
#define WPB_SZ (16 * 512)     // 8 kt x 2 nt fragments
#define WND_NFRAG 28          // 2 layers x 7 kt x 2 nt
#define WPD_SZ (WND_NFRAG * 512)
#define WIH_NFRAG 320         // 10 p x 4 q x 4 wv x 2 kt
#define WIH_SZ (WIH_NFRAG * 512)

// ---- fast activations: gfx1250 hardware tanh ------------------------------
#if __has_builtin(__builtin_amdgcn_tanhf)
__device__ __forceinline__ float tanh_fast(float x) {
    return __builtin_amdgcn_tanhf(x);      // v_tanh_f32 (TRANS32)
}
#else
__device__ __forceinline__ float tanh_fast(float x) { return tanhf(x); }
#endif

__device__ __forceinline__ float sigmoid_fast(float x) {
    // sig(x) = 0.5 * (1 + tanh(x/2)) -> 1 TRANS + 2 VALU, branch-free
    return 0.5f + 0.5f * tanh_fast(0.5f * x);
}

__device__ __forceinline__ v8f wmma16x16x32(v16h a, v16h b, v8f c) {
    // (neg_a, A, neg_b, B, c_mod, C, reuse_a, reuse_b)
    return __builtin_amdgcn_wmma_f32_16x16x32_f16(false, a, false, b,
                                                  (short)0, c, false, false);
}

// A-fragment (16x32 f16) local K index for vector element j, lane-half hi.
// lanes 0-15 hold K in {0..15}; lanes 16-31 hold K in {16..31}.
__device__ __forceinline__ int a_klocal(int j, int hi) {
    return j + 16 * hi;
}

// ===========================================================================
// Prep: swizzle row-major [KT*32][32] f32 weights into f16 B-fragments.
// Fragment f = kt*2 + nt holds element j of lane l at dst[f*512 + l*16 + j]
// with n = nt*16 + (l&15), k = kt*32 + 16*(l>>4) + j.
// ===========================================================================
__global__ __launch_bounds__(256)
void swizzle_b32(const float* __restrict__ src, _Float16* __restrict__ dst,
                 int nfrag)
{
    const int e = blockIdx.x * 256 + threadIdx.x;
    if (e >= nfrag * 512) return;
    const int f = e >> 9, r = e & 511;
    const int kt = f >> 1, nt = f & 1;
    const int lane = r >> 4, j = r & 15;
    const int n = nt * 16 + (lane & 15);
    const int k = kt * 32 + 16 * (lane >> 4) + j;
    dst[e] = (_Float16)src[k * 32 + n];
}

// LSTM weights: src W[p][n_abs][k] ([P,256,64]); B[k][n_abs] = W[n_abs][k].
// Fragment f = ((p*4+q)*4+wv)*2+kt, n_abs = 64q + 16wv + (l&15),
// k = 32kt + 16*(l>>4) + j.
__global__ __launch_bounds__(256)
void swizzle_lstm(const float* __restrict__ src, _Float16* __restrict__ dst)
{
    const int e = blockIdx.x * 256 + threadIdx.x;
    if (e >= WIH_NFRAG * 512) return;
    const int f = e >> 9, r = e & 511;
    const int kt = f & 1, wv = (f >> 1) & 3, q = (f >> 3) & 3, p = f >> 5;
    const int lane = r >> 4, j = r & 15;
    const int n_abs = 64 * q + 16 * wv + (lane & 15);
    const int k = 32 * kt + 16 * (lane >> 4) + j;
    dst[e] = (_Float16)src[((size_t)p * 256 + n_abs) * 64 + k];
}

// ===========================================================================
// Kernel 1: signed GCN. grid = (B*T)/K1_WAVES blocks of K1_WAVES*32 threads.
// ===========================================================================
__global__ __launch_bounds__(K1_WAVES * 32)
void sgcn_kernel(const float* __restrict__ x,
                 const float* __restrict__ Apos,
                 const float* __restrict__ Aneg,
                 const float* __restrict__ bpb, const float* __restrict__ bnb,
                 const float* __restrict__ bpd, const float* __restrict__ bnd,
                 const _Float16* __restrict__ wpbf,
                 const _Float16* __restrict__ wnbf,
                 const _Float16* __restrict__ wpdf,
                 const _Float16* __restrict__ wndf,
                 _Float16* __restrict__ z)
{
    // Per-wave scratch. sAcat columns:
    //   base layer:  [0,128) agg_pos | [128,256) agg_neg | [256,384) x (f16)
    //   deep layers: [0,128) 4 agg blocks | [128,160) h_pos | [160,192) h_neg
    __shared__ float    sAdj[K1_WAVES][2][PNUM][PNUM];
    __shared__ _Float16 sAcat[K1_WAVES][16][384];

    const int wv   = threadIdx.x >> 5;
    const int lane = threadIdx.x & 31;
    const int inst = blockIdx.x * K1_WAVES + wv;
    const int b = inst / TSTEPS;
    const int t = inst % TSTEPS;

    const int n  = lane & 15;     // WMMA N-col / D-col
    const int hi = lane >> 4;     // lane half

    // ---- 1) row-normalized signed adjacencies ----------------------------
    if (lane < 20) {
        const int s = lane / PNUM, i = lane % PNUM;
        const float* Ar = (s ? Aneg : Apos)
                        + ((size_t)(b * TSTEPS + t)) * PNUM * PNUM + i * PNUM;
        float row[PNUM], sum = 0.0f;
        #pragma unroll
        for (int j = 0; j < PNUM; ++j) { row[j] = Ar[j]; sum += row[j]; }
        const float inv = 1.0f / (sum + EPSV);
        #pragma unroll
        for (int j = 0; j < PNUM; ++j) sAdj[wv][s][i][j] = row[j] * inv;
    }

    // ---- 2) stage x (f16, float4-coalesced) into cols [256,384) ----------
    const float4* xin4 = (const float4*)(x + ((size_t)(b * TSTEPS + t)) * PNUM * DFEAT);
    for (int idx = lane; idx < 16 * (DFEAT / 4); idx += 32) {
        const int m = idx >> 5, c4 = idx & 31;
        float4 v = make_float4(0.f, 0.f, 0.f, 0.f);
        if (m < PNUM) v = xin4[m * (DFEAT / 4) + c4];
        v4h hv;
        hv[0] = (_Float16)v.x; hv[1] = (_Float16)v.y;
        hv[2] = (_Float16)v.z; hv[3] = (_Float16)v.w;
        *(v4h*)&sAcat[wv][m][256 + 4 * c4] = hv;
    }
    __syncthreads();

    // ---- 3) base aggregation: agg_s = An_s @ x -> cols [s*128, s*128+128) -
    for (int idx = lane; idx < 2 * 16 * (DFEAT / 4); idx += 32) {
        const int s = idx >> 9, rem = idx & 511;
        const int m = rem >> 5, c4 = (rem & 31) * 4;
        float a0 = 0.f, a1 = 0.f, a2 = 0.f, a3 = 0.f;
        if (m < PNUM) {
            #pragma unroll
            for (int j = 0; j < PNUM; ++j) {
                const v4h xv = *(const v4h*)&sAcat[wv][j][256 + c4];
                const float w = sAdj[wv][s][m][j];
                a0 += w * (float)xv[0]; a1 += w * (float)xv[1];
                a2 += w * (float)xv[2]; a3 += w * (float)xv[3];
            }
        }
        v4h ov;
        ov[0] = (_Float16)a0; ov[1] = (_Float16)a1;
        ov[2] = (_Float16)a2; ov[3] = (_Float16)a3;
        *(v4h*)&sAcat[wv][m][s * 128 + c4] = ov;
    }
    __syncthreads();

    // ---- 4) base dense layers: tanh([agg_s | x] @ W_s_base + b_s_base) ----
    float hcur[2][16];                       // [sign][ntile*8 + r]
    #pragma unroll
    for (int s = 0; s < 2; ++s) {
        const _Float16* Wf = s ? wnbf : wpbf;
        const float*  bias = s ? bnb  : bpb;
        #pragma unroll
        for (int nt = 0; nt < 2; ++nt) {
            v8f acc = {};
            #pragma unroll
            for (int kt = 0; kt < 8; ++kt) {
                const int c0 = (kt < 4) ? (s * 128 + 32 * kt)
                                        : (256 + 32 * (kt - 4));
                v16h a;
                #pragma unroll
                for (int j = 0; j < 16; ++j)
                    a[j] = sAcat[wv][n][c0 + a_klocal(j, hi)];
                const v16h bw = *(const v16h*)(Wf + (size_t)(kt * 2 + nt) * 512
                                               + lane * 16);
                acc = wmma16x16x32(a, bw, acc);
            }
            #pragma unroll
            for (int r = 0; r < 8; ++r)
                hcur[s][nt * 8 + r] = tanh_fast(acc[r] + bias[nt * 16 + n]);
        }
    }
    // store h_pos -> cols [128,160), h_neg -> cols [160,192)
    #pragma unroll
    for (int s = 0; s < 2; ++s)
        #pragma unroll
        for (int nt = 0; nt < 2; ++nt)
            #pragma unroll
            for (int r = 0; r < 8; ++r)
                sAcat[wv][r + 8 * hi][128 + 32 * s + nt * 16 + n] =
                    (_Float16)hcur[s][nt * 8 + r];
    __syncthreads();

    // ---- 5) two deep layers ----------------------------------------------
    #pragma unroll
    for (int layer = 0; layer < 2; ++layer) {
        // 4 aggregation blocks -> cols [0,128):
        //   bi=0: An_p@h_pos  bi=1: An_p@h_neg  bi=2: An_n@h_pos  bi=3: An_n@h_neg
        for (int idx = lane; idx < 4 * 16 * 8; idx += 32) {
            const int bi = idx >> 7, rem = idx & 127;
            const int m = rem >> 3, c4 = (rem & 7) * 4;
            const int sa = bi >> 1, sh = bi & 1;
            float a0 = 0.f, a1 = 0.f, a2 = 0.f, a3 = 0.f;
            if (m < PNUM) {
                #pragma unroll
                for (int j = 0; j < PNUM; ++j) {
                    const v4h hv = *(const v4h*)&sAcat[wv][j][128 + 32 * sh + c4];
                    const float w = sAdj[wv][sa][m][j];
                    a0 += w * (float)hv[0]; a1 += w * (float)hv[1];
                    a2 += w * (float)hv[2]; a3 += w * (float)hv[3];
                }
            }
            v4h ov;
            ov[0] = (_Float16)a0; ov[1] = (_Float16)a1;
            ov[2] = (_Float16)a2; ov[3] = (_Float16)a3;
            *(v4h*)&sAcat[wv][m][32 * bi + c4] = ov;
        }
        __syncthreads();

        // concat7 @ W_deep: k-tiles 0..5 = cols [0,192); k-tile 6 reuses h_s
        #pragma unroll
        for (int s = 0; s < 2; ++s) {
            const _Float16* Wf = s ? wndf : wpdf;
            const float*  bias = (s ? bnd : bpd) + layer * 32;
            #pragma unroll
            for (int nt = 0; nt < 2; ++nt) {
                v8f acc = {};
                #pragma unroll
                for (int kt = 0; kt < 7; ++kt) {
                    const int c0 = (kt < 6) ? (32 * kt) : (128 + 32 * s);
                    v16h a;
                    #pragma unroll
                    for (int j = 0; j < 16; ++j)
                        a[j] = sAcat[wv][n][c0 + a_klocal(j, hi)];
                    const size_t f = ((size_t)layer * 7 + kt) * 2 + nt;
                    const v16h bw = *(const v16h*)(Wf + f * 512 + lane * 16);
                    acc = wmma16x16x32(a, bw, acc);
                }
                #pragma unroll
                for (int r = 0; r < 8; ++r)
                    hcur[s][nt * 8 + r] = tanh_fast(acc[r] + bias[nt * 16 + n]);
            }
        }
        #pragma unroll
        for (int s = 0; s < 2; ++s)
            #pragma unroll
            for (int nt = 0; nt < 2; ++nt)
                #pragma unroll
                for (int r = 0; r < 8; ++r)
                    sAcat[wv][r + 8 * hi][128 + 32 * s + nt * 16 + n] =
                        (_Float16)hcur[s][nt * 8 + r];
        __syncthreads();
    }

    // ---- 6) z = [h_pos | h_neg] (f16) -> workspace -----------------------
    _Float16* zo = z + ((size_t)(b * TSTEPS + t)) * PNUM * 64;
    #pragma unroll
    for (int s = 0; s < 2; ++s)
        #pragma unroll
        for (int nt = 0; nt < 2; ++nt)
            #pragma unroll
            for (int r = 0; r < 8; ++r) {
                const int m = r + 8 * hi;
                if (m < PNUM)
                    zo[m * 64 + s * 32 + nt * 16 + n] =
                        (_Float16)hcur[s][nt * 8 + r];
            }
}

// ===========================================================================
// Kernel 2: per-player LSTM. grid = P * (B/16) blocks of 128 threads.
// Wave wv owns hidden columns [16*wv, 16*wv+16) of a 16-batch-row tile.
// ===========================================================================
__global__ __launch_bounds__(128)
void lstm_kernel(const _Float16* __restrict__ z,
                 const _Float16* __restrict__ wihf,
                 const _Float16* __restrict__ whhf,
                 const float* __restrict__ bih,
                 const float* __restrict__ bhh,
                 float* __restrict__ out)
{
    __shared__ _Float16 sH[16][HIDN + 8];    // recurrent h (f16), padded

    const int wv   = threadIdx.x >> 5;
    const int lane = threadIdx.x & 31;
    const int p  = blockIdx.x / (BGAMES / 16);
    const int b0 = (blockIdx.x % (BGAMES / 16)) * 16;

    const int n  = lane & 15;
    const int hi = lane >> 4;

    // ---- hoist weight B-fragments into registers (vector loads) ----------
    v16h wih[4][2], whh[4][2];
    float bias[4];
    #pragma unroll
    for (int q = 0; q < 4; ++q) {
        const int n0 = 64 * q + 16 * wv;     // gate q, this wave's 16 hid cols
        bias[q] = bih[p * 256 + n0 + n] + bhh[p * 256 + n0 + n];
        #pragma unroll
        for (int kt = 0; kt < 2; ++kt) {
            const size_t f = (size_t)(((p * 4 + q) * 4 + wv) * 2 + kt);
            wih[q][kt] = *(const v16h*)(wihf + f * 512 + lane * 16);
            whh[q][kt] = *(const v16h*)(whhf + f * 512 + lane * 16);
        }
    }

    // ---- init state -------------------------------------------------------
    float c[8];
    #pragma unroll
    for (int r = 0; r < 8; ++r) c[r] = 0.0f;
    for (int idx = threadIdx.x; idx < 16 * (HIDN + 8); idx += 128)
        sH[idx / (HIDN + 8)][idx % (HIDN + 8)] = (_Float16)0.0f;
    __syncthreads();

    // ---- recurrence over vote steps --------------------------------------
    for (int t = 0; t < TSTEPS; ++t) {
        // A fragments: z_t (global f16) and h (LDS f16)
        v16h az[2], ah[2];
        const _Float16* zrow =
            z + ((((size_t)(b0 + n)) * TSTEPS + t) * PNUM + p) * 64;
        #pragma unroll
        for (int kt = 0; kt < 2; ++kt)
            #pragma unroll
            for (int j = 0; j < 16; ++j) {
                const int kl = a_klocal(j, hi) + 32 * kt;
                az[kt][j] = zrow[kl];
                ah[kt][j] = sH[n][kl];
            }
        if (t + 1 < TSTEPS)
            __builtin_prefetch(zrow + PNUM * 64, 0, 1);  // next step's z slice

        // gates: i,f,g,o for this wave's (16 batch x 16 hid) patch
        float g4[4][8];
        #pragma unroll
        for (int q = 0; q < 4; ++q) {
            v8f acc = {};
            acc = wmma16x16x32(az[0], wih[q][0], acc);
            acc = wmma16x16x32(az[1], wih[q][1], acc);
            acc = wmma16x16x32(ah[0], whh[q][0], acc);
            acc = wmma16x16x32(ah[1], whh[q][1], acc);
            #pragma unroll
            for (int r = 0; r < 8; ++r) g4[q][r] = acc[r] + bias[q];
        }
        __syncthreads();                     // all waves done reading old sH

        #pragma unroll
        for (int r = 0; r < 8; ++r) {
            const float ig = sigmoid_fast(g4[0][r]);
            const float fg = sigmoid_fast(g4[1][r]);
            const float gg = tanh_fast(g4[2][r]);
            const float og = sigmoid_fast(g4[3][r]);
            c[r] = fg * c[r] + ig * gg;
            const float hval = og * tanh_fast(c[r]);
            const int m = r + 8 * hi;
            sH[m][16 * wv + n] = (_Float16)hval;
            out[(((size_t)(b0 + m)) * TSTEPS + t) * PNUM * HIDN
                + p * HIDN + 16 * wv + n] = hval;
        }
        __syncthreads();                     // new h visible for next step
    }
}

// ===========================================================================
extern "C" void kernel_launch(void* const* d_in, const int* in_sizes, int n_in,
                              void* d_out, int out_size, void* d_ws, size_t ws_size,
                              hipStream_t stream) {
    (void)in_sizes; (void)n_in; (void)out_size; (void)ws_size;
    const float* x    = (const float*)d_in[0];
    const float* Apos = (const float*)d_in[1];
    const float* Aneg = (const float*)d_in[2];
    const float* Wpb  = (const float*)d_in[3];
    const float* bpb  = (const float*)d_in[4];
    const float* Wnb  = (const float*)d_in[5];
    const float* bnb  = (const float*)d_in[6];
    const float* Wpd  = (const float*)d_in[7];
    const float* bpd  = (const float*)d_in[8];
    const float* Wnd  = (const float*)d_in[9];
    const float* bnd  = (const float*)d_in[10];
    const float* Wih  = (const float*)d_in[11];
    const float* Whh  = (const float*)d_in[12];
    const float* bih  = (const float*)d_in[13];
    const float* bhh  = (const float*)d_in[14];

    // workspace carve-up (f16 elements)
    _Float16* zws  = (_Float16*)d_ws;        // [B,T,P,64] f16 = 21 MB
    _Float16* wpbf = zws  + Z_SZ;
    _Float16* wnbf = wpbf + WPB_SZ;
    _Float16* wpdf = wnbf + WPB_SZ;
    _Float16* wndf = wpdf + WPD_SZ;
    _Float16* wihf = wndf + WPD_SZ;
    _Float16* whhf = wihf + WIH_SZ;

    // 1) one-time (per launch) weight swizzle into WMMA B-fragment layout
    swizzle_b32 <<<16 * 2,        256, 0, stream>>>(Wpb, wpbf, 16);
    swizzle_b32 <<<16 * 2,        256, 0, stream>>>(Wnb, wnbf, 16);
    swizzle_b32 <<<WND_NFRAG * 2, 256, 0, stream>>>(Wpd, wpdf, WND_NFRAG);
    swizzle_b32 <<<WND_NFRAG * 2, 256, 0, stream>>>(Wnd, wndf, WND_NFRAG);
    swizzle_lstm<<<WIH_NFRAG * 2, 256, 0, stream>>>(Wih, wihf);
    swizzle_lstm<<<WIH_NFRAG * 2, 256, 0, stream>>>(Whh, whhf);

    // 2) SGCN over all (b,t) graph instances
    sgcn_kernel<<<(BGAMES * TSTEPS) / K1_WAVES, K1_WAVES * 32, 0, stream>>>(
        x, Apos, Aneg, bpb, bnb, bpd, bnd, wpbf, wnbf, wpdf, wndf, zws);

    // 3) per-player LSTM recurrence
    lstm_kernel<<<PNUM * (BGAMES / 16), 128, 0, stream>>>(
        zws, wihf, whhf, bih, bhh, (float*)d_out);
}
